// GraphAttention_37349035606451
// MI455X (gfx1250) — matrix operand order repaired and checked by reading, not compiled, not run on hardware
//
#include <hip/hip_runtime.h>
#include <hip/hip_bf16.h>
#include <cstddef>

typedef __attribute__((ext_vector_type(16))) _Float16 v16h;
typedef __attribute__((ext_vector_type(8)))  _Float16 v8h;
typedef __attribute__((ext_vector_type(4)))  _Float16 v4h;
typedef __attribute__((ext_vector_type(8)))  float    v8f;
typedef __attribute__((ext_vector_type(4)))  float    v4f;
typedef __attribute__((ext_vector_type(4)))  int      v4i;

#define N_NODES 8192
#define F_IN    256
#define F_OUT   128

// ---------------------------------------------------------------------------
// Elementwise converts: h -> f16 (row major), W -> f16 transposed (128x256)
// ---------------------------------------------------------------------------
__global__ __launch_bounds__(256) void cvt_h_kernel(const float* __restrict__ h,
                                                    _Float16* __restrict__ h16) {
    int idx = blockIdx.x * 256 + threadIdx.x;   // 8192*256 total
    h16[idx] = (_Float16)h[idx];
}

__global__ __launch_bounds__(256) void cvt_w_kernel(const float* __restrict__ W,
                                                    _Float16* __restrict__ w16t) {
    int idx = blockIdx.x * 256 + threadIdx.x;   // 256*128 total
    int k = idx >> 7;        // 0..255
    int n = idx & 127;       // 0..127
    w16t[n * F_IN + k] = (_Float16)W[k * F_OUT + n];
}

// ---------------------------------------------------------------------------
// Wh = h @ W via WMMA f32_16x16x32_f16.
// Block = 16 rows (iBase), 8 waves, wave w owns cols [16w, 16w+16).
// Writes Wh (f32, 8192x128) and WhT16 (f16, 128x8192).
// ---------------------------------------------------------------------------
__global__ __launch_bounds__(256) void wh_gemm_kernel(const _Float16* __restrict__ h16,
                                                      const _Float16* __restrict__ w16t,
                                                      float* __restrict__ wh,
                                                      _Float16* __restrict__ wht16) {
    const int lane  = threadIdx.x & 31;
    const int wave  = threadIdx.x >> 5;
    const int iBase = blockIdx.x * 16;
    const int nBase = wave * 16;
    const int row   = lane & 15;        // A: M row / B: N col owned by this lane
    const int hiL   = lane >> 4;        // lane half select
    const int hb    = hiL * 16;         // A fragment K offset (halfs)
    const int kb    = hiL * 16;         // B fragment K offset (halfs)

    v8f c = {};
    const _Float16* arow = h16  + (size_t)(iBase + row) * F_IN;
    const _Float16* bcol = w16t + (size_t)(nBase + row) * F_IN;

    for (int k0 = 0; k0 < F_IN; k0 += 32) {
        // A 16x32 f16 fragment: lane holds row M=row, K = k0+hb+{0..15}
        v8h a0 = *(const v8h*)(arow + k0 + hb);
        v8h a1 = *(const v8h*)(arow + k0 + hb + 8);
        v16h A = __builtin_shufflevector(a0, a1, 0,1,2,3,4,5,6,7,8,9,10,11,12,13,14,15);
        // B 32x16 f16 fragment: lane holds col N=row, K = k0+kb+{0..15} (contig in W^T)
        v16h B = *(const v16h*)(bcol + k0 + kb);
        c = __builtin_amdgcn_wmma_f32_16x16x32_f16(false, A, false, B,
                                                   (short)0, c, false, false);
    }

#pragma unroll
    for (int r = 0; r < 8; ++r) {
        int m  = r + hiL * 8;
        int gi = iBase + m;
        int gn = nBase + row;
        float v = c[r];
        wh[(size_t)gi * F_OUT + gn]       = v;
        wht16[(size_t)gn * N_NODES + gi]  = (_Float16)v;
    }
}

// ---------------------------------------------------------------------------
// s1[i] = dot(Wh[i,:], a[0:128]);  s2[i] = dot(Wh[i,:], a[128:256])
// ---------------------------------------------------------------------------
__global__ __launch_bounds__(256) void s_kernel(const float* __restrict__ wh,
                                                const float* __restrict__ a,
                                                float* __restrict__ s1,
                                                float* __restrict__ s2) {
    int i = blockIdx.x * 256 + threadIdx.x;     // 0..8191
    const float* r = wh + (size_t)i * F_OUT;
    float d1 = 0.f, d2 = 0.f;
#pragma unroll 4
    for (int k = 0; k < F_OUT; ++k) {
        float v = r[k];
        d1 += v * a[k];
        d2 += v * a[F_OUT + k];
    }
    s1[i] = d1;
    s2[i] = d2;
}

// ---------------------------------------------------------------------------
// Pass 1: rowm[i] = leaky_relu(s1[i] + max_{j: adj[i,j]>0} s2[j])
// (leaky_relu is monotonic, so this equals the masked row-max of e.)
// ---------------------------------------------------------------------------
__global__ __launch_bounds__(256) void rowmax_kernel(const int* __restrict__ adj,
                                                     const float* __restrict__ s1,
                                                     const float* __restrict__ s2,
                                                     float* __restrict__ rowm) {
    __shared__ float red[256];
    const int i = blockIdx.x;
    const int* arow = adj + (size_t)i * N_NODES;
    float m = -3.0e38f;
    for (int j = threadIdx.x * 4; j < N_NODES; j += 1024) {
        __builtin_prefetch(arow + j + 4096, 0, 0);
        v4i av = *(const v4i*)(arow + j);
        v4f sv = *(const v4f*)(s2 + j);
#pragma unroll
        for (int q = 0; q < 4; ++q) {
            if (av[q] > 0) {
                float v = sv[q];
                m = m > v ? m : v;
            }
        }
    }
    red[threadIdx.x] = m;
    __syncthreads();
    for (int off = 128; off > 0; off >>= 1) {
        if (threadIdx.x < off)
            red[threadIdx.x] = fmaxf(red[threadIdx.x], red[threadIdx.x + off]);
        __syncthreads();
    }
    if (threadIdx.x == 0) {
        float M = red[0];
        float e = s1[i] + M;
        e = e > 0.f ? e : 0.2f * e;
        rowm[i] = (M < -1.0e38f) ? 0.f : e;   // isolated row guard
    }
}

// ---------------------------------------------------------------------------
// Pass 2 (fused): out[i,:] = (sum_j P[i,j] * Wh[j,:]) / Z[i]
// P[i,j] = adj>0 ? exp(lrelu(s1[i]+s2[j]) - rowm[i]) : 0, built 16x64-tile-
// by-tile in LDS (f16, double buffered) and fed into 2 WMMAs per barrier.
// Producer: thread t fills 4 consecutive cols of row t>>4 (one int4 adj load,
// one float4 s2 load, one 8B LDS store per tile).
// Block = 16 rows; wave w owns output cols [16w,16w+16); K loop over 8192.
// ---------------------------------------------------------------------------
#define PT_LDH 72   // padded LDS row stride in halfs (144B: 16B aligned)

__global__ __launch_bounds__(256) void attn_gemm_kernel(const int* __restrict__ adj,
                                                        const float* __restrict__ s1,
                                                        const float* __restrict__ s2,
                                                        const float* __restrict__ rowm,
                                                        const _Float16* __restrict__ wht16,
                                                        float* __restrict__ out) {
    __shared__ __align__(16) _Float16 Pt[2][16 * PT_LDH];
    __shared__ float Zl[16];
    __shared__ float s1l[16];
    __shared__ float rml[16];

    const int t     = threadIdx.x;
    const int lane  = t & 31;
    const int wave  = t >> 5;
    const int iBase = blockIdx.x * 16;

    if (t < 16) {
        s1l[t] = s1[iBase + t];
        rml[t] = rowm[iBase + t];
        Zl[t]  = 0.f;
    }
    __syncthreads();

    // P-tile producer role: row prow, columns [pcol, pcol+4)
    const int prow = t >> 4;            // 0..15
    const int pcol = (t & 15) * 4;      // 0..60
    const int* adjrow = adj + (size_t)(iBase + prow) * N_NODES + pcol;
    float zacc = 0.f;

    // WMMA consumer role
    const int row = lane & 15;
    const int hiL = lane >> 4;
    const int hb  = hiL * 16;
    const int kb  = hiL * 16;
    const int nBase = wave * 16;
    const _Float16* bcol = wht16 + (size_t)(nBase + row) * N_NODES;

    v8f c = {};
    int buf = 0;

    for (int k0 = 0; k0 < N_NODES; k0 += 64) {
        // --- produce 16x64 probability tile (4 consecutive entries/thread) ---
        __builtin_prefetch(adjrow + k0 + 2048, 0, 0);
        v4i av = *(const v4i*)(adjrow + k0);
        v4f sv = *(const v4f*)(s2 + k0 + pcol);
        const float s1r = s1l[prow];
        const float mr  = rml[prow];
        v4h pv;
#pragma unroll
        for (int q = 0; q < 4; ++q) {
            float e = s1r + sv[q];
            e = e > 0.f ? e : 0.2f * e;
            float p = (av[q] > 0) ? __expf(e - mr) : 0.f;
            pv[q] = (_Float16)p;
            zacc += p;
        }
        *(v4h*)(&Pt[buf][prow * PT_LDH + pcol]) = pv;
        __syncthreads();

        // --- consume: 2 WMMAs; A from LDS tile, B from transposed Wh ---
        const _Float16* arow = &Pt[buf][row * PT_LDH];

        v8h a0 = *(const v8h*)(arow + hb);
        v8h a1 = *(const v8h*)(arow + hb + 8);
        v16h A0 = __builtin_shufflevector(a0, a1, 0,1,2,3,4,5,6,7,8,9,10,11,12,13,14,15);
        v16h B0 = *(const v16h*)(bcol + k0 + kb);
        c = __builtin_amdgcn_wmma_f32_16x16x32_f16(false, A0, false, B0,
                                                   (short)0, c, false, false);

        v8h a2 = *(const v8h*)(arow + 32 + hb);
        v8h a3 = *(const v8h*)(arow + 32 + hb + 8);
        v16h A1 = __builtin_shufflevector(a2, a3, 0,1,2,3,4,5,6,7,8,9,10,11,12,13,14,15);
        v16h B1 = *(const v16h*)(bcol + k0 + 32 + kb);
        c = __builtin_amdgcn_wmma_f32_16x16x32_f16(false, A1, false, B1,
                                                   (short)0, c, false, false);
        buf ^= 1;
    }

    atomicAdd(&Zl[prow], zacc);
    __syncthreads();

#pragma unroll
    for (int r = 0; r < 8; ++r) {
        int m = r + hiL * 8;
        float z = Zl[m];
        out[(size_t)(iBase + m) * F_OUT + nBase + row] = c[r] / z;
    }
}

// ---------------------------------------------------------------------------
// Launch
// ---------------------------------------------------------------------------
extern "C" void kernel_launch(void* const* d_in, const int* in_sizes, int n_in,
                              void* d_out, int out_size, void* d_ws, size_t ws_size,
                              hipStream_t stream) {
    const float* h   = (const float*)d_in[0];
    const int*   adj = (const int*)d_in[1];
    const float* W   = (const float*)d_in[2];
    const float* a   = (const float*)d_in[3];
    float*       out = (float*)d_out;

    char* ws = (char*)d_ws;
    _Float16* h16   = (_Float16*)ws;  ws += (size_t)N_NODES * F_IN * sizeof(_Float16);
    _Float16* w16t  = (_Float16*)ws;  ws += (size_t)F_IN * F_OUT * sizeof(_Float16);
    float*    wh    = (float*)ws;     ws += (size_t)N_NODES * F_OUT * sizeof(float);
    _Float16* wht16 = (_Float16*)ws;  ws += (size_t)N_NODES * F_OUT * sizeof(_Float16);
    float*    s1    = (float*)ws;     ws += (size_t)N_NODES * sizeof(float);
    float*    s2    = (float*)ws;     ws += (size_t)N_NODES * sizeof(float);
    float*    rowm  = (float*)ws;     ws += (size_t)N_NODES * sizeof(float);

    cvt_h_kernel<<<(N_NODES * F_IN) / 256, 256, 0, stream>>>(h, h16);
    cvt_w_kernel<<<(F_IN * F_OUT) / 256, 256, 0, stream>>>(W, w16t);
    wh_gemm_kernel<<<N_NODES / 16, 256, 0, stream>>>(h16, w16t, wh, wht16);
    s_kernel<<<N_NODES / 256, 256, 0, stream>>>(wh, a, s1, s2);
    rowmax_kernel<<<N_NODES, 256, 0, stream>>>(adj, s1, s2, rowm);
    attn_gemm_kernel<<<N_NODES / 16, 256, 0, stream>>>(adj, s1, s2, rowm, wht16, out);
}
